// MultiHeadedAttention_41489384079621
// MI455X (gfx1250) — compile-verified, matrix operand order checked by reading, not run on hardware
//
#include <hip/hip_runtime.h>
#include <hip/hip_bf16.h>

#define SEQ 2048
#define DIM 1024
#define NH  16
#define HD  64

typedef __attribute__((ext_vector_type(16))) __bf16        bf16x16;
typedef __attribute__((ext_vector_type(8)))  float         f32x8;
typedef __attribute__((ext_vector_type(4)))  float         f32x4;
typedef __attribute__((ext_vector_type(4)))  unsigned int  u32x4;

__device__ __forceinline__ unsigned short f2bf(float f) {
    unsigned int u = __float_as_uint(f);
    u += 0x7FFFu + ((u >> 16) & 1u);          // round-to-nearest-even
    return (unsigned short)(u >> 16);
}
__device__ __forceinline__ float bf2f(unsigned short h) {
    return __uint_as_float(((unsigned int)h) << 16);
}
__device__ __forceinline__ unsigned int pack2(float lo, float hi) {
    return (unsigned int)f2bf(lo) | ((unsigned int)f2bf(hi) << 16);
}

// Load a 16x32 bf16 fragment (A-layout; also B-layout when fed B^T row-major).
// Lane l (<16): row l, K = 0..7 and 16..23 ; lane l+16: row l, K = 8..15 and 24..31.
__device__ __forceinline__ bf16x16 frag_from_bf(const unsigned short* tile, int stride) {
    int lane = threadIdx.x & 31;
    int r  = lane & 15;
    int kk = (lane >> 4) << 3;
    const unsigned short* rp = tile + r * stride;
    union { u32x4 q[2]; bf16x16 v; } cv;
    cv.q[0] = *(const u32x4*)(rp + kk);
    cv.q[1] = *(const u32x4*)(rp + 16 + kk);
    return cv.v;
}

__device__ __forceinline__ f32x8 wmma_bf16(bf16x16 a, bf16x16 b, f32x8 c) {
    return __builtin_amdgcn_wmma_f32_16x16x32_bf16(false, a, false, b, (short)0, c, false, false);
}

__device__ __forceinline__ float half_reduce_max(float v) {
    v = fmaxf(v, __shfl_xor(v, 1, 32));
    v = fmaxf(v, __shfl_xor(v, 2, 32));
    v = fmaxf(v, __shfl_xor(v, 4, 32));
    v = fmaxf(v, __shfl_xor(v, 8, 32));
    return v;
}
__device__ __forceinline__ float half_reduce_sum(float v) {
    v += __shfl_xor(v, 1, 32);
    v += __shfl_xor(v, 2, 32);
    v += __shfl_xor(v, 4, 32);
    v += __shfl_xor(v, 8, 32);
    return v;
}

// LDS byte address for async-load destinations: on AMDGCN the low 32 bits of a
// generic pointer into the LDS aperture are the wave-relative LDS offset.
__device__ __forceinline__ unsigned lds_addr(const void* p) {
    return (unsigned)(unsigned long long)p;
}

// ---------------------------------------------------------------------------
// Kernel 0: one-shot fp32 -> bf16 weight conversion (1024x1024 per launch)
// ---------------------------------------------------------------------------
__global__ __launch_bounds__(256) void w2bf_kernel(const float* __restrict__ src,
                                                   unsigned short* __restrict__ dst) {
    int idx = (blockIdx.x * 256 + threadIdx.x) * 4;
    f32x4 f = *(const f32x4*)(src + idx);
    unsigned int* d = (unsigned int*)(dst + idx);
    d[0] = pack2(f.x, f.y);
    d[1] = pack2(f.z, f.w);
}

// ---------------------------------------------------------------------------
// Kernel 1: projection  out = x @ W^T + b   (fp32 act, bf16 weights, bf16 out)
// vtranspose==0 -> out[b][h][s][dk]   (Q, K)
// vtranspose==1 -> out[b][h][dk][s]   (V, pre-transposed for P*V B-operand)
// grid 2048 (256 m-tiles x 8 n-groups), block 256 (8 waves x 16 cols)
// ---------------------------------------------------------------------------
__global__ __launch_bounds__(256) void proj_kernel(const float* __restrict__ x,
                                                   const unsigned short* __restrict__ Wbf,
                                                   const float* __restrict__ bias,
                                                   unsigned short* __restrict__ out,
                                                   int vtranspose) {
    int blk  = blockIdx.x;
    int m0   = (blk >> 3) << 4;
    int tid  = threadIdx.x;
    int wave = tid >> 5;
    int lane = tid & 31;
    int n0   = ((blk & 7) << 7) + (wave << 4);

    __shared__ unsigned short xs[16 * DIM];      // 32 KB activation tile, bf16

    const float* xg = x + (size_t)m0 * DIM;
    for (int c = tid; c < 4096; c += 256) {      // 16 rows * 256 float4 chunks
        int row = c >> 8;
        int col = (c & 255) << 2;
        f32x4 f = *(const f32x4*)(xg + row * DIM + col);
        unsigned int* dst = (unsigned int*)(xs + row * DIM + col);
        dst[0] = pack2(f.x, f.y);
        dst[1] = pack2(f.z, f.w);
    }
    __syncthreads();

    f32x8 acc = {};
    const unsigned short* wrow = Wbf + (size_t)n0 * DIM;
    for (int c = 0; c < 32; c++) {
        __builtin_prefetch(wrow + (c + 1) * 32, 0, 1);
        bf16x16 a  = frag_from_bf(xs + c * 32, DIM);
        bf16x16 bf = frag_from_bf(wrow + c * 32, DIM);
        acc = wmma_bf16(a, bf, acc);
    }

    int ncol = lane & 15, mrow = (lane >> 4) << 3;
    int n  = n0 + ncol;
    float bn = bias[n];
    int h = n >> 6, dk = n & 63;
#pragma unroll
    for (int v = 0; v < 8; v++) {
        int m = m0 + mrow + v;
        int b = m >> 11, s = m & (SEQ - 1);
        float val = acc[v] + bn;
        size_t idx = vtranspose
            ? (((size_t)b * NH + h) * HD + dk) * SEQ + s
            : (((size_t)b * NH + h) * SEQ + s) * HD + dk;
        out[idx] = f2bf(val);
    }
}

// ---------------------------------------------------------------------------
// Kernel 2: RoPE on q,k head buffers [B,H,S,DK] (bf16 in/out)
// ---------------------------------------------------------------------------
__global__ __launch_bounds__(256) void rope_kernel(unsigned short* __restrict__ q,
                                                   unsigned short* __restrict__ k) {
    int idx  = blockIdx.x * 256 + threadIdx.x;
    int i    = idx & 31;
    int srow = idx >> 5;                  // b*H*S + h*S + s
    int s    = srow & (SEQ - 1);
    size_t base = (size_t)srow * HD + i;

    float inv = __expf(-(float)i * (9.210340371976184f / 32.0f)); // 10000^(-i/32)
    float ang = (float)s * inv;
    float sn, cs;
    __sincosf(ang, &sn, &cs);

    float q1 = bf2f(q[base]), q2 = bf2f(q[base + 32]);
    q[base]      = f2bf(q1 * cs - q2 * sn);
    q[base + 32] = f2bf(q2 * cs + q1 * sn);

    float k1 = bf2f(k[base]), k2 = bf2f(k[base + 32]);
    k[base]      = f2bf(k1 * cs - k2 * sn);
    k[base + 32] = f2bf(k2 * cs + k1 * sn);
}

// ---------------------------------------------------------------------------
// Kernel 3: causal flash attention.
// grid = B*H*(S/64) = 1024 workgroups, 4 waves each; wave owns 16 q rows.
// K tile (32x64) and V^T tile (64x32) staged into LDS with CDNA5 async loads.
// ---------------------------------------------------------------------------
__global__ __launch_bounds__(128) void attn_kernel(const unsigned short* __restrict__ qh,
                                                   const unsigned short* __restrict__ kh,
                                                   const unsigned short* __restrict__ vT,
                                                   unsigned short* __restrict__ attn) {
    int blk  = blockIdx.x;
    int tile = blk & 31;                  // S/64 = 32 q-tiles per (b,h)
    int bh   = blk >> 5;
    int q0wg = tile * 64;
    int tid  = threadIdx.x;
    int wave = tid >> 5;
    int lane = tid & 31;
    int q0   = q0wg + wave * 16;

    __shared__ unsigned short k_lds[32 * HD];        // 4 KB
    __shared__ unsigned short v_lds[HD * 32];        // 4 KB (dk-major)
    __shared__ unsigned short p_lds[4 * 16 * 32];    // 4 KB, per-wave P scratch

    const unsigned short* qg = qh + ((size_t)bh * SEQ + q0) * HD;
    bf16x16 aq0 = frag_from_bf(qg, HD);          // K dims 0..31
    bf16x16 aq1 = frag_from_bf(qg + 32, HD);     // K dims 32..63

    f32x8 o0 = {}, o1 = {}, o2 = {}, o3 = {};
    float mi[8], li[8];
#pragma unroll
    for (int v = 0; v < 8; v++) { mi[v] = -3.0e38f; li[v] = 0.0f; }

    const unsigned short* kbase = kh + (size_t)bh * SEQ * HD;
    const unsigned short* vbase = vT + (size_t)bh * HD * SEQ;

    int mrow = (lane >> 4) << 3;
    int ncol = lane & 15;

    // per-thread staging addresses (constant across the j-loop)
    unsigned kdst = lds_addr(k_lds + tid * 16);
    unsigned koff = (unsigned)(tid * 16) * 2u;                  // bytes
    int vrow = tid >> 1, vc0 = (tid & 1) * 16;
    unsigned vdst = lds_addr(v_lds + vrow * 32 + vc0);
    unsigned voff = (unsigned)(vrow * SEQ + vc0) * 2u;          // bytes

    for (int j = 0; j < q0wg + 64; j += 32) {
        { // async staging: global -> LDS, no VGPR round-trip (ASYNCcnt-tracked)
            const unsigned short* kg = kbase + (size_t)j * HD;  // contiguous 32x64
            const unsigned short* vg = vbase + j;               // 64 rows, stride SEQ
            asm volatile("global_load_async_to_lds_b128 %0, %1, %2"
                         :: "v"(kdst), "v"(koff), "s"(kg) : "memory");
            asm volatile("global_load_async_to_lds_b128 %0, %1, %2 offset:16"
                         :: "v"(kdst), "v"(koff), "s"(kg) : "memory");
            asm volatile("global_load_async_to_lds_b128 %0, %1, %2"
                         :: "v"(vdst), "v"(voff), "s"(vg) : "memory");
            asm volatile("global_load_async_to_lds_b128 %0, %1, %2 offset:16"
                         :: "v"(vdst), "v"(voff), "s"(vg) : "memory");
            asm volatile("s_wait_asynccnt 0" ::: "memory");
        }
        __syncthreads();

        if (j < q0 + 16) {   // tile not fully above this wave's diagonal
            f32x8 s0 = {}, s1 = {};
            s0 = wmma_bf16(aq0, frag_from_bf(k_lds,            HD), s0);
            s0 = wmma_bf16(aq1, frag_from_bf(k_lds + 32,       HD), s0);
            s1 = wmma_bf16(aq0, frag_from_bf(k_lds + 16 * HD,      HD), s1);
            s1 = wmma_bf16(aq1, frag_from_bf(k_lds + 16 * HD + 32, HD), s1);

            unsigned short* pw = p_lds + wave * 512;
#pragma unroll
            for (int v = 0; v < 8; v++) {
                int qgidx = q0 + mrow + v;
                int key0  = j + ncol;
                float sv0 = (key0      <= qgidx) ? s0[v] * 0.125f : -1.0e9f;
                float sv1 = (key0 + 16 <= qgidx) ? s1[v] * 0.125f : -1.0e9f;
                float tm   = half_reduce_max(fmaxf(sv0, sv1));
                float mnew = fmaxf(mi[v], tm);
                float p0 = __expf(sv0 - mnew);
                float p1 = __expf(sv1 - mnew);
                float rs = half_reduce_sum(p0 + p1);
                float alpha = __expf(mi[v] - mnew);
                li[v] = li[v] * alpha + rs;
                mi[v] = mnew;
                o0[v] *= alpha; o1[v] *= alpha; o2[v] *= alpha; o3[v] *= alpha;
                pw[(mrow + v) * 32 + ncol]      = f2bf(p0);
                pw[(mrow + v) * 32 + ncol + 16] = f2bf(p1);
            }
            asm volatile("s_wait_dscnt 0" ::: "memory");   // P store -> A-layout reload
            bf16x16 pa = frag_from_bf(pw, 32);
            o0 = wmma_bf16(pa, frag_from_bf(v_lds,           32), o0);
            o1 = wmma_bf16(pa, frag_from_bf(v_lds + 16 * 32, 32), o1);
            o2 = wmma_bf16(pa, frag_from_bf(v_lds + 32 * 32, 32), o2);
            o3 = wmma_bf16(pa, frag_from_bf(v_lds + 48 * 32, 32), o3);
        }
        __syncthreads();
    }

    int b = bh >> 4, h = bh & 15;
#pragma unroll
    for (int v = 0; v < 8; v++) {
        float invl = 1.0f / li[v];
        int s = q0 + mrow + v;
        size_t base = ((size_t)b * SEQ + s) * DIM + h * HD + ncol;
        attn[base + 0]  = f2bf(o0[v] * invl);
        attn[base + 16] = f2bf(o1[v] * invl);
        attn[base + 32] = f2bf(o2[v] * invl);
        attn[base + 48] = f2bf(o3[v] * invl);
    }
}

// ---------------------------------------------------------------------------
// Kernel 4: output projection  y = attn(bf16) @ Wo^T + bo  -> fp32 d_out
// ---------------------------------------------------------------------------
__global__ __launch_bounds__(256) void outproj_kernel(const unsigned short* __restrict__ attn,
                                                      const unsigned short* __restrict__ Wobf,
                                                      const float* __restrict__ bo,
                                                      float* __restrict__ out) {
    int blk  = blockIdx.x;
    int m0   = (blk >> 3) << 4;
    int tid  = threadIdx.x;
    int wave = tid >> 5;
    int lane = tid & 31;
    int n0   = ((blk & 7) << 7) + (wave << 4);

    __shared__ unsigned short xs[16 * DIM];
    const unsigned short* ag = attn + (size_t)m0 * DIM;
    for (int c = tid; c < 2048; c += 256) {        // 16K bf16 = 2048 x 16B
        *(u32x4*)(xs + c * 8) = *(const u32x4*)(ag + c * 8);
    }
    __syncthreads();

    f32x8 acc = {};
    const unsigned short* wrow = Wobf + (size_t)n0 * DIM;
    for (int c = 0; c < 32; c++) {
        __builtin_prefetch(wrow + (c + 1) * 32, 0, 1);
        bf16x16 a  = frag_from_bf(xs + c * 32, DIM);
        bf16x16 bf = frag_from_bf(wrow + c * 32, DIM);
        acc = wmma_bf16(a, bf, acc);
    }

    int ncol = lane & 15, mrow = (lane >> 4) << 3;
    int n = n0 + ncol;
    float bn = bo[n];
#pragma unroll
    for (int v = 0; v < 8; v++) {
        out[(size_t)(m0 + mrow + v) * DIM + n] = acc[v] + bn;
    }
}

// ---------------------------------------------------------------------------
extern "C" void kernel_launch(void* const* d_in, const int* in_sizes, int n_in,
                              void* d_out, int out_size, void* d_ws, size_t ws_size,
                              hipStream_t stream) {
    const float* query = (const float*)d_in[0];
    const float* key   = (const float*)d_in[1];
    const float* value = (const float*)d_in[2];
    // d_in[3] = causal mask: handled analytically, not read.
    const float* Wq = (const float*)d_in[4];
    const float* bq = (const float*)d_in[5];
    const float* Wk = (const float*)d_in[6];
    const float* bk = (const float*)d_in[7];
    const float* Wv = (const float*)d_in[8];
    const float* bv = (const float*)d_in[9];
    const float* Wo = (const float*)d_in[10];
    const float* bo = (const float*)d_in[11];

    const size_t HEAD_ELEMS = (size_t)2 * NH * SEQ * HD;  // 4,194,304
    const size_t W_ELEMS    = (size_t)DIM * DIM;          // 1,048,576
    unsigned short* qh  = (unsigned short*)d_ws;
    unsigned short* khp = qh  + HEAD_ELEMS;
    unsigned short* vTp = khp + HEAD_ELEMS;
    unsigned short* atn = vTp + HEAD_ELEMS;
    unsigned short* wqb = atn + HEAD_ELEMS;
    unsigned short* wkb = wqb + W_ELEMS;
    unsigned short* wvb = wkb + W_ELEMS;
    unsigned short* wob = wvb + W_ELEMS;

    w2bf_kernel<<<1024, 256, 0, stream>>>(Wq, wqb);
    w2bf_kernel<<<1024, 256, 0, stream>>>(Wk, wkb);
    w2bf_kernel<<<1024, 256, 0, stream>>>(Wv, wvb);
    w2bf_kernel<<<1024, 256, 0, stream>>>(Wo, wob);

    proj_kernel<<<2048, 256, 0, stream>>>(query, wqb, bq, qh,  0);
    proj_kernel<<<2048, 256, 0, stream>>>(key,   wkb, bk, khp, 0);
    proj_kernel<<<2048, 256, 0, stream>>>(value, wvb, bv, vTp, 1);
    rope_kernel<<<8192, 256, 0, stream>>>(qh, khp);
    attn_kernel<<<1024, 128, 0, stream>>>(qh, khp, vTp, atn);
    outproj_kernel<<<2048, 256, 0, stream>>>(atn, wob, bo, (float*)d_out);
}
